// MOELayer_31542239822189
// MI455X (gfx1250) — matrix operand order repaired and checked by reading, not compiled, not run on hardware
//
#include <hip/hip_runtime.h>
#include <hip/hip_bf16.h>

// ---------------- types for WMMA ----------------
typedef __attribute__((ext_vector_type(16))) __bf16        v16bf;
typedef __attribute__((ext_vector_type(8)))  float         v8f;
typedef __attribute__((ext_vector_type(4)))  unsigned int  v4u;

// gcc-style vector matching the async-LDS builtin's pointer pointee type
typedef int moe_v4i __attribute__((vector_size(16)));

union FragB { v16bf v; v4u q[2]; };

// fp32 -> bf16 round-to-nearest-even
__device__ __forceinline__ unsigned short f2bf(float f) {
    unsigned int u = __float_as_uint(f);
    unsigned int r = (u + 0x7fffu + ((u >> 16) & 1u)) >> 16;
    return (unsigned short)r;
}

// ---------------- MoE problem constants ----------------
#define MOE_E   8
#define MOE_K   2
#define MOE_N   8192      // B*S = 4*2048 tokens
#define MOE_M   1024
#define MOE_H   4096
#define MOE_CAP 2048      // K*N/E

// ---------------- CDNA5 async global->LDS path ----------------
#if defined(__has_builtin)
#if __has_builtin(__builtin_amdgcn_global_load_async_to_lds_b128) && \
    __has_builtin(__builtin_amdgcn_s_wait_asynccnt)
#define MOE_USE_ASYNC_LDS 1
#endif
#endif

#ifdef MOE_USE_ASYNC_LDS
// generic LDS pointer -> addrspace(3): low 32 bits of the generic address are
// the LDS offset (flat aperture encoding); AS(1) is identity for global.
__device__ __forceinline__ void async_copy16(const void* g, void* lds) {
    __builtin_amdgcn_global_load_async_to_lds_b128(
        (__attribute__((address_space(1))) moe_v4i*)(uintptr_t)g,
        (__attribute__((address_space(3))) moe_v4i*)(unsigned int)(uintptr_t)lds,
        0, 0);
}
__device__ __forceinline__ void async_wait0() {
    __builtin_amdgcn_s_wait_asynccnt(0);
}
#endif

// =======================================================
// 1) fp32 [R,C] -> bf16 [C,R] transpose+convert (per expert via blockIdx.z)
// =======================================================
__global__ void transpose_f32_to_bf16(const float* __restrict__ src,
                                      unsigned short* __restrict__ dst,
                                      int R, int C) {
    __shared__ float tile[32][33];
    const size_t eoff = (size_t)blockIdx.z * (size_t)R * (size_t)C;
    const int c0 = blockIdx.x * 32, r0 = blockIdx.y * 32;
    const int tx = threadIdx.x, ty = threadIdx.y; // 32 x 8
#pragma unroll
    for (int i = 0; i < 32; i += 8)
        tile[ty + i][tx] = src[eoff + (size_t)(r0 + ty + i) * C + (c0 + tx)];
    __syncthreads();
#pragma unroll
    for (int i = 0; i < 32; i += 8)
        dst[eoff + (size_t)(c0 + ty + i) * R + (r0 + tx)] = f2bf(tile[tx][ty + i]);
}

// =======================================================
// 2) gate: one wave32 per token. logits = x[n,:] @ wg[M,8], softmax, top-2
// =======================================================
__global__ void gate_kernel(const float* __restrict__ x,
                            const float* __restrict__ wg,
                            int* __restrict__ topi,
                            float* __restrict__ gate) {
    const int wave = threadIdx.x >> 5;
    const int lane = threadIdx.x & 31;
    const int n = blockIdx.x * 8 + wave;
    const float* xr = x + (size_t)n * MOE_M;

    float acc[8];
#pragma unroll
    for (int e = 0; e < 8; ++e) acc[e] = 0.f;

    for (int m = lane; m < MOE_M; m += 32) {
        float xv = xr[m];
        const float4* w4 = (const float4*)(wg + (size_t)m * 8);
        float4 a = w4[0], b = w4[1];
        acc[0] += xv * a.x; acc[1] += xv * a.y; acc[2] += xv * a.z; acc[3] += xv * a.w;
        acc[4] += xv * b.x; acc[5] += xv * b.y; acc[6] += xv * b.z; acc[7] += xv * b.w;
    }
#pragma unroll
    for (int off = 16; off > 0; off >>= 1) {
#pragma unroll
        for (int e = 0; e < 8; ++e) acc[e] += __shfl_down(acc[e], off, 32);
    }
    if (lane == 0) {
        float mx = acc[0];
#pragma unroll
        for (int e = 1; e < 8; ++e) mx = fmaxf(mx, acc[e]);
        float p[8], s = 0.f;
#pragma unroll
        for (int e = 0; e < 8; ++e) { p[e] = __expf(acc[e] - mx); s += p[e]; }
        float inv = 1.f / s;
#pragma unroll
        for (int e = 0; e < 8; ++e) p[e] *= inv;
        int i1 = 0; float v1 = p[0];
#pragma unroll
        for (int e = 1; e < 8; ++e) if (p[e] > v1) { v1 = p[e]; i1 = e; }
        int i2 = -1; float v2 = -1.f;
#pragma unroll
        for (int e = 0; e < 8; ++e) if (e != i1 && p[e] > v2) { v2 = p[e]; i2 = e; }
        float rs = 1.f / (v1 + v2);
        topi[n * 2 + 0] = i1; topi[n * 2 + 1] = i2;
        gate[n * 2 + 0] = v1 * rs; gate[n * 2 + 1] = v2 * rs;
    }
}

// =======================================================
// 3) slot-major capacity scan: single wave32, ballot prefix per expert
// =======================================================
__global__ void scan_kernel(const int* __restrict__ topi,
                            float* __restrict__ gate,
                            int* __restrict__ pos,
                            int capacity) {
    const int lane = threadIdx.x;
    int cnt[8];
#pragma unroll
    for (int e = 0; e < 8; ++e) cnt[e] = 0;
    const unsigned int lt = (1u << lane) - 1u;
    const int total = 2 * MOE_N;
    for (int j0 = 0; j0 < total; j0 += 32) {
        int j = j0 + lane;
        int k = j / MOE_N;
        int n = j - k * MOE_N;
        int idx = n * 2 + k;
        int e = topi[idx];
        int mypos = 0;
#pragma unroll
        for (int q = 0; q < 8; ++q) {
            unsigned int m = (unsigned int)__ballot(e == q);
            if (e == q) mypos = cnt[q] + __popc(m & lt);
            cnt[q] += __popc(m);
        }
        bool valid = mypos < capacity;
        pos[idx] = valid ? mypos : -1;
        if (!valid) gate[idx] = 0.0f;
    }
}

// =======================================================
// 4a) zero bf16 dispatch buffer (16B granules)
// =======================================================
__global__ void zero16_kernel(v4u* __restrict__ p, size_t n16) {
    size_t i = (size_t)blockIdx.x * blockDim.x + threadIdx.x;
    if (i < n16) { v4u z = {0u, 0u, 0u, 0u}; p[i] = z; }
}

// 4b) dispatch scatter: token (n,k) -> disp[e][pos][:] as bf16
__global__ void dispatch_kernel(const float* __restrict__ x,
                                const int* __restrict__ topi,
                                const int* __restrict__ pos,
                                unsigned short* __restrict__ disp) {
    const int j = blockIdx.x;
    const int p = pos[j];
    if (p < 0) return;
    const int e = topi[j];
    const int n = j >> 1;
    const float* src = x + (size_t)n * MOE_M;
    unsigned short* dst = disp + ((size_t)e * MOE_CAP + p) * MOE_M;
    int i = threadIdx.x * 8;
    float4 a = *(const float4*)(src + i);
    float4 b = *(const float4*)(src + i + 4);
    v4u o;
    o.x = (unsigned int)f2bf(a.x) | ((unsigned int)f2bf(a.y) << 16);
    o.y = (unsigned int)f2bf(a.z) | ((unsigned int)f2bf(a.w) << 16);
    o.z = (unsigned int)f2bf(b.x) | ((unsigned int)f2bf(b.y) << 16);
    o.w = (unsigned int)f2bf(b.z) | ((unsigned int)f2bf(b.w) << 16);
    *(v4u*)(dst + i) = o;
}

// =======================================================
// 5) bf16 WMMA GEMM:  Out[z] = A[z] (Crows x K) * Bt[z]^T (Ncols x K) + bias
//    128x128 block tile, 8 waves, each wave 2x4 tiles of 16x16,
//    K-step 32 (v_wmma_f32_16x16x32_bf16), double-buffered LDS,
//    tile fill via GLOBAL_LOAD_ASYNC_TO_LDS_B128 (ASYNCcnt) when available.
// =======================================================
template <bool RELU_BF16_OUT>
__global__ __launch_bounds__(256, 1)
void gemm_bf16_tn(const unsigned short* __restrict__ A,
                  const unsigned short* __restrict__ Bt,
                  const float* __restrict__ bias,
                  void* __restrict__ Out,
                  int Crows, int Ncols, int K) {
    __shared__ unsigned short sA[2][128][40];
    __shared__ unsigned short sB[2][128][40];

    const int z = blockIdx.z;
    const unsigned short* Ae = A  + (size_t)z * Crows * K;
    const unsigned short* Be = Bt + (size_t)z * Ncols * K;
    const float*          be = bias + (size_t)z * Ncols;

    const int row0 = blockIdx.y * 128;
    const int col0 = blockIdx.x * 128;

    const int t    = threadIdx.x;
    const int lr   = t >> 1;          // 0..127: tile row (A) / tile col (B)
    const int lc   = (t & 1) * 16;    // 0 or 16: K-chunk within 32
    const int wave = t >> 5, lane = t & 31;
    const int wr   = (wave >> 1) * 32;  // wave row base within block tile
    const int wc   = (wave & 1) * 64;   // wave col base within block tile
    const int l15  = lane & 15, lh = lane >> 4;

    // per-thread global tile-fill source (advance by k along the row)
    const unsigned short* gA = Ae + (size_t)(row0 + lr) * K + lc;
    const unsigned short* gB = Be + (size_t)(col0 + lr) * K + lc;

    v8f zero = {0.f, 0.f, 0.f, 0.f, 0.f, 0.f, 0.f, 0.f};
    v8f acc[2][4];
#pragma unroll
    for (int i = 0; i < 2; ++i)
#pragma unroll
        for (int u = 0; u < 4; ++u) acc[i][u] = zero;

    const int KT = K >> 5;
    int buf = 0;

#ifdef MOE_USE_ASYNC_LDS
    // -------- async DMA tile-fill pipeline (no VGPR staging) --------
    async_copy16(gA, &sA[0][lr][lc]);
    async_copy16(gB, &sB[0][lr][lc]);
    async_wait0();
    __syncthreads();

    for (int kt = 0; kt < KT; ++kt) {
        if (kt + 1 < KT) {
            const int kn = (kt + 1) << 5;
            async_copy16(gA + kn, &sA[buf ^ 1][lr][lc]);
            async_copy16(gB + kn, &sB[buf ^ 1][lr][lc]);
        }
#else
    // -------- fallback: global->VGPR->LDS double buffering --------
    v4u ra = *(const v4u*)gA;
    v4u rb = *(const v4u*)gB;
    *(v4u*)&sA[0][lr][lc] = ra;
    *(v4u*)&sB[0][lr][lc] = rb;
    __syncthreads();

    for (int kt = 0; kt < KT; ++kt) {
        if (kt + 1 < KT) {
            const int kn = (kt + 1) << 5;
            ra = *(const v4u*)(gA + kn);
            rb = *(const v4u*)(gB + kn);
        }
#endif
        // A fragments: 16-bit A 16x32 layout — lanes 0-15: K{0..7,16..23},
        // lanes 16-31: K{8..15,24..31}, row = lane&15
        FragB af[2], bfm[4];
#pragma unroll
        for (int i = 0; i < 2; ++i) {
            const int r = wr + i * 16 + l15;
            af[i].q[0] = *(const v4u*)&sA[buf][r][lh * 8];
            af[i].q[1] = *(const v4u*)&sA[buf][r][16 + lh * 8];
        }
        // B fragments: 32x16 — lanes 0-15 hold K{0..7,16..23} of col (lane&15),
        // lanes 16-31 hold K{8..15,24..31} (Bt stored [col][K])
#pragma unroll
        for (int u = 0; u < 4; ++u) {
            const int c = wc + u * 16 + l15;
            bfm[u].q[0] = *(const v4u*)&sB[buf][c][lh * 8];
            bfm[u].q[1] = *(const v4u*)&sB[buf][c][16 + lh * 8];
        }
#pragma unroll
        for (int i = 0; i < 2; ++i)
#pragma unroll
            for (int u = 0; u < 4; ++u)
                acc[i][u] = __builtin_amdgcn_wmma_f32_16x16x32_bf16(
                    false, af[i].v, false, bfm[u].v, (short)0, acc[i][u],
                    false, false);

#ifdef MOE_USE_ASYNC_LDS
        if (kt + 1 < KT) async_wait0();   // own async stores into buf^1 done
        __syncthreads();                  // everyone done reading buf / filling buf^1
        buf ^= 1;
    }
#else
        if (kt + 1 < KT) {
            *(v4u*)&sA[buf ^ 1][lr][lc] = ra;
            *(v4u*)&sB[buf ^ 1][lr][lc] = rb;
        }
        __syncthreads();
        buf ^= 1;
    }
#endif

    // C/D layout: VGPR i -> row i (lanes 0-15) / row i+8 (lanes 16-31), col = lane&15
#pragma unroll
    for (int i = 0; i < 2; ++i) {
#pragma unroll
        for (int u = 0; u < 4; ++u) {
            const int colg = col0 + wc + u * 16 + l15;
            const float bv = be[colg];
#pragma unroll
            for (int r8 = 0; r8 < 8; ++r8) {
                const int rowg = row0 + wr + i * 16 + (lane < 16 ? r8 : r8 + 8);
                float v = acc[i][u][r8] + bv;
                if (RELU_BF16_OUT) {
                    v = v > 0.f ? v : 0.f;
                    ((unsigned short*)Out)[((size_t)z * Crows + rowg) * Ncols + colg] = f2bf(v);
                } else {
                    ((float*)Out)[((size_t)z * Crows + rowg) * Ncols + colg] = v;
                }
            }
        }
    }
}

// =======================================================
// 6) combine: out[n,:] = sum_k gate[n,k] * y[e_k, pos_k, :]
// =======================================================
__global__ void combine_kernel(const float* __restrict__ y,
                               const int* __restrict__ topi,
                               const int* __restrict__ pos,
                               const float* __restrict__ gate,
                               float* __restrict__ out) {
    const int n = blockIdx.x;
    const int m = threadIdx.x * 4;
    float r0 = 0.f, r1 = 0.f, r2 = 0.f, r3 = 0.f;
#pragma unroll
    for (int k = 0; k < 2; ++k) {
        const float g = gate[n * 2 + k];
        const int p = pos[n * 2 + k];
        if (g != 0.f && p >= 0) {
            const int e = topi[n * 2 + k];
            const float4 v = *(const float4*)(y + ((size_t)e * MOE_CAP + p) * MOE_M + m);
            r0 += g * v.x; r1 += g * v.y; r2 += g * v.z; r3 += g * v.w;
        }
    }
    float4 o; o.x = r0; o.y = r1; o.z = r2; o.w = r3;
    *(float4*)(out + (size_t)n * MOE_M + m) = o;
}

// =======================================================
// host launcher
// =======================================================
extern "C" void kernel_launch(void* const* d_in, const int* in_sizes, int n_in,
                              void* d_out, int out_size, void* d_ws, size_t ws_size,
                              hipStream_t stream) {
    (void)in_sizes; (void)n_in; (void)out_size; (void)ws_size;
    const float* x  = (const float*)d_in[0];  // [N, M]
    const float* wg = (const float*)d_in[1];  // [M, E]
    const float* w1 = (const float*)d_in[2];  // [E, M, H]
    const float* b1 = (const float*)d_in[3];  // [E, H]
    const float* w2 = (const float*)d_in[4];  // [E, H, M]
    const float* b2 = (const float*)d_in[5];  // [E, M]
    float* out = (float*)d_out;               // [N, M]

    char* ws = (char*)d_ws;
    size_t o = 0;
    auto take = [&](size_t bytes) -> void* {
        void* p = ws + o;
        o += (bytes + 255) & ~(size_t)255;
        return p;
    };
    unsigned short* w1t  = (unsigned short*)take((size_t)MOE_E * MOE_H * MOE_M * 2); // [E,H,M] bf16
    unsigned short* w2t  = (unsigned short*)take((size_t)MOE_E * MOE_M * MOE_H * 2); // [E,M,H] bf16
    unsigned short* disp = (unsigned short*)take((size_t)MOE_E * MOE_CAP * MOE_M * 2);
    unsigned short* hbuf = (unsigned short*)take((size_t)MOE_E * MOE_CAP * MOE_H * 2);
    float*          ybuf = (float*)take((size_t)MOE_E * MOE_CAP * MOE_M * 4);
    int*            topi = (int*)take((size_t)MOE_N * 2 * 4);
    float*          gate = (float*)take((size_t)MOE_N * 2 * 4);
    int*            pos  = (int*)take((size_t)MOE_N * 2 * 4);

    // 1) weight convert+transpose to bf16 (column-of-B contiguous along K)
    transpose_f32_to_bf16<<<dim3(MOE_H / 32, MOE_M / 32, MOE_E), dim3(32, 8), 0, stream>>>(
        w1, w1t, MOE_M, MOE_H);
    transpose_f32_to_bf16<<<dim3(MOE_M / 32, MOE_H / 32, MOE_E), dim3(32, 8), 0, stream>>>(
        w2, w2t, MOE_H, MOE_M);

    // 2) gate
    gate_kernel<<<MOE_N / 8, 256, 0, stream>>>(x, wg, topi, gate);

    // 3) slot-major scan (single wave)
    scan_kernel<<<1, 32, 0, stream>>>(topi, gate, pos, MOE_CAP);

    // 4) zero + dispatch
    {
        size_t n16 = (size_t)MOE_E * MOE_CAP * MOE_M * 2 / 16;
        zero16_kernel<<<(unsigned)((n16 + 255) / 256), 256, 0, stream>>>((v4u*)disp, n16);
    }
    dispatch_kernel<<<MOE_N * 2, 128, 0, stream>>>(x, topi, pos, disp);

    // 5) expert FFN: h = relu(disp @ w1 + b1) ; y = h @ w2 + b2
    gemm_bf16_tn<true><<<dim3(MOE_H / 128, MOE_CAP / 128, MOE_E), 256, 0, stream>>>(
        disp, w1t, b1, (void*)hbuf, MOE_CAP, MOE_H, MOE_M);
    gemm_bf16_tn<false><<<dim3(MOE_M / 128, MOE_CAP / 128, MOE_E), 256, 0, stream>>>(
        hbuf, w2t, b2, (void*)ybuf, MOE_CAP, MOE_M, MOE_H);

    // 6) combine
    combine_kernel<<<MOE_N, 256, 0, stream>>>(ybuf, topi, pos, gate, out);
}
